// MultiheadedAttention_55740085567718
// MI455X (gfx1250) — matrix operand order repaired and checked by reading, not compiled, not run on hardware
//
#include <hip/hip_runtime.h>
#include <hip/hip_bf16.h>

typedef _Float16 v8h  __attribute__((ext_vector_type(8)));
typedef _Float16 v16h __attribute__((ext_vector_type(16)));
typedef float    v8f  __attribute__((ext_vector_type(8)));

#define B_ 8
#define H_ 8
#define T_ 1024
#define S_ 64
#define E_ 512
#define NEG_ (-1.0e9f)
#define INV_SQRT_E 0.04419417382415922f  // 1/sqrt(512)

#define WAIT_DS() asm volatile("s_wait_dscnt 0" ::: "memory")

static __device__ inline v16h mk16(v8h lo, v8h hi) {
  v16h r;
#pragma unroll
  for (int i = 0; i < 8; ++i) { r[i] = lo[i]; r[i + 8] = hi[i]; }
  return r;
}

// A-matrix fragment (16x32 f16 tile, row-major source).  ISA 7.12.2:
// lane holds row (lane&15); K elems {h8..h8+7} and {16+h8..16+h8+7}, h8=(lane>>4)*8
static __device__ inline v16h frag_a(const _Float16* rowp, int half8) {
  v8h lo = *(const v8h*)(rowp + half8);
  v8h hi = *(const v8h*)(rowp + 16 + half8);
  return mk16(lo, hi);
}

// B-matrix fragment, loading B^T (NxK) row-major: lane holds B^T row (lane&15),
// K elems {h8..h8+7} and {16+h8..16+h8+7}, h8=halfk/2=(lane>>4)*8
static __device__ inline v16h frag_b(const _Float16* rowp, int halfk) {
  v8h lo = *(const v8h*)(rowp + (halfk >> 1));
  v8h hi = *(const v8h*)(rowp + 16 + (halfk >> 1));
  return mk16(lo, hi);
}

static __device__ inline v8f wmma_f16(v16h a, v16h b, v8f c) {
  return __builtin_amdgcn_wmma_f32_16x16x32_f16(false, a, false, b, (short)0, c,
                                                false, false);
}

// ---------------------------------------------------------------- projections
// q[b,h,t,o] = sum_s x[b,t,h*64+s]*Wq[o,s]  (same for k); v stored transposed.
__global__ __launch_bounds__(256) void proj_qkv_kernel(
    const float* __restrict__ x, const float* __restrict__ Wq,
    const float* __restrict__ Wk, const float* __restrict__ Wv,
    _Float16* __restrict__ q16, _Float16* __restrict__ k16,
    _Float16* __restrict__ vT16) {
  __shared__ float xs[E_];
  int blk = blockIdx.x;           // flat (b,t)
  int b = blk >> 10, t = blk & (T_ - 1);
  const float* xp = x + (size_t)blk * E_;
  for (int i = threadIdx.x; i < E_; i += 256) xs[i] = xp[i];
  __syncthreads();
  for (int oid = threadIdx.x; oid < 3 * E_; oid += 256) {
    int mat = oid / E_;
    int rem = oid - mat * E_;
    int h = rem >> 6, o = rem & 63;
    const float* W = (mat == 0) ? Wq : ((mat == 1) ? Wk : Wv);
    const float* wrow = W + o * S_;
    const float* xrow = xs + h * S_;
    float acc = 0.f;
#pragma unroll
    for (int s = 0; s < S_; ++s) acc = fmaf(xrow[s], wrow[s], acc);
    int bh = b * H_ + h;
    if (mat == 0)       q16[((size_t)bh * T_ + t) * S_ + o] = (_Float16)acc;
    else if (mat == 1)  k16[((size_t)bh * T_ + t) * S_ + o] = (_Float16)acc;
    else                vT16[((size_t)bh * S_ + o) * T_ + t] = (_Float16)acc;
  }
}

__global__ __launch_bounds__(256) void f32_to_f16_kernel(
    const float* __restrict__ in, _Float16* __restrict__ out, int n) {
  int i = blockIdx.x * 256 + threadIdx.x;
  if (i < n) out[i] = (_Float16)in[i];
}

// mean of v over T for masked query rows (softmax of constant row = uniform)
__global__ __launch_bounds__(64) void vmean_kernel(
    const _Float16* __restrict__ vT16, float* __restrict__ vmean) {
  int bh = blockIdx.x, s = threadIdx.x;
  const _Float16* p = vT16 + ((size_t)bh * S_ + s) * T_;
  float acc = 0.f;
  for (int t = 0; t < T_; ++t) acc += (float)p[t];
  vmean[bh * S_ + s] = acc * (1.0f / T_);
}

// -------------------------------------------------------- flash attention
// one wave (32 threads) per (b,h, 16-query tile); streams 32 keys / iter
__global__ __launch_bounds__(32) void flash_attn_kernel(
    const _Float16* __restrict__ q16, const _Float16* __restrict__ k16,
    const _Float16* __restrict__ vT16, const _Float16* __restrict__ er16,
    const float* __restrict__ vmean, const int* __restrict__ mask,
    _Float16* __restrict__ attn16) {
  int bh = blockIdx.x;
  int b = bh >> 3, h = bh & 7;
  int t0 = blockIdx.y * 16;
  int lane = threadIdx.x;
  int l16 = lane & 15;
  int half8 = (lane >> 4) << 3;
  int halfk = (lane >> 4) << 4;

  __shared__ float   Rbuf[16][48];   // rel band R[t', n-n_base]
  __shared__ _Float16 Pbuf[16][32];  // P tile C-layout -> A-layout transpose

  const _Float16* qb = q16 + (size_t)bh * T_ * S_;
  const _Float16* kb = k16 + (size_t)bh * T_ * S_;
  const _Float16* vb = vT16 + (size_t)bh * S_ * T_;
  const _Float16* eb = er16 + (size_t)h * T_ * S_;

  // Q fragments (16 rows x 64 K, two K-steps), reused all iterations
  const _Float16* qr = qb + (t0 + l16) * S_;
  v16h qa0 = frag_a(qr, half8);
  v16h qa1 = frag_a(qr + 32, half8);

  v8f O[4] = {};                 // 16 x 64 output accumulator (4 N-tiles)
  float rmax[8], rsum[8];
#pragma unroll
  for (int v = 0; v < 8; ++v) { rmax[v] = -3.0e38f; rsum[v] = 0.f; }

  int tlast = t0 + 15;
  for (int m0 = 0; m0 <= tlast; m0 += 32) {
    if (m0 + 32 < T_) {          // pull next K tile toward the caches
      __builtin_prefetch(kb + (size_t)(m0 + 32 + l16) * S_, 0, 1);
    }
    // ---- S_qk = Q K^T  (two 16-key N tiles, K=64 via 2 wmma each)
    v8f sj[2];
#pragma unroll
    for (int j = 0; j < 2; ++j) {
      const _Float16* kr = kb + (size_t)(m0 + j * 16 + l16) * S_;
      v8f c = {};
      c = wmma_f16(qa0, frag_b(kr, halfk), c);
      c = wmma_f16(qa1, frag_b(kr + 32, halfk), c);
      sj[j] = c;
    }
    // ---- rel band R[t', n] for n = n_base + [0,48), n = m - t + (T-1)
    int nbase = m0 - t0 + (T_ - 16);
#pragma unroll
    for (int rt = 0; rt < 3; ++rt) {
      int n = nbase + rt * 16 + l16;
      n = (n > T_ - 1) ? (T_ - 1) : n;      // clamped rows are causally masked
      const _Float16* er = eb + (size_t)n * S_;
      v8f c = {};
      c = wmma_f16(qa0, frag_b(er, halfk), c);
      c = wmma_f16(qa1, frag_b(er + 32, halfk), c);
#pragma unroll
      for (int v = 0; v < 8; ++v) Rbuf[half8 + v][rt * 16 + l16] = c[v];
    }
    WAIT_DS();

    // ---- assemble scores, online softmax, stage P (f16) in LDS
#pragma unroll
    for (int v = 0; v < 8; ++v) {
      int tq = half8 + v;                       // query row within tile
      int c0 = l16 - tq + 15;                   // band col for m' = l16
      float s0 = sj[0][v] * INV_SQRT_E + Rbuf[tq][c0];
      float s1 = sj[1][v] * INV_SQRT_E + Rbuf[tq][c0 + 16];
      s0 = (m0 + l16 > t0 + tq) ? NEG_ : s0;          // causal
      s1 = (m0 + 16 + l16 > t0 + tq) ? NEG_ : s1;
      float mx = fmaxf(s0, s1);
#pragma unroll
      for (int d = 1; d < 16; d <<= 1) mx = fmaxf(mx, __shfl_xor(mx, d, 32));
      float nm = fmaxf(rmax[v], mx);
      float alpha = __expf(rmax[v] - nm);
      rmax[v] = nm;
      float p0 = __expf(s0 - nm);
      float p1 = __expf(s1 - nm);
      float ps = p0 + p1;
#pragma unroll
      for (int d = 1; d < 16; d <<= 1) ps += __shfl_xor(ps, d, 32);
      rsum[v] = rsum[v] * alpha + ps;
#pragma unroll
      for (int nt = 0; nt < 4; ++nt) O[nt][v] *= alpha;
      Pbuf[tq][l16] = (_Float16)p0;
      Pbuf[tq][16 + l16] = (_Float16)p1;
    }
    WAIT_DS();

    // ---- O += P V   (A = P 16x32 from LDS, B^T = vT rows, K = 32 keys)
    v8h plo = *(const v8h*)(&Pbuf[l16][half8]);
    v8h phi = *(const v8h*)(&Pbuf[l16][16 + half8]);
    v16h pa = mk16(plo, phi);
#pragma unroll
    for (int nt = 0; nt < 4; ++nt) {
      const _Float16* vr = vb + (size_t)(nt * 16 + l16) * T_ + m0;
      O[nt] = wmma_f16(pa, frag_b(vr, halfk), O[nt]);
    }
  }

  // ---- epilogue: normalize; masked query rows -> mean(v); store f16 [b,t,e]
#pragma unroll
  for (int nt = 0; nt < 4; ++nt) {
    int sc = nt * 16 + l16;
    float vm = vmean[bh * S_ + sc];
#pragma unroll
    for (int v = 0; v < 8; ++v) {
      int t = t0 + half8 + v;
      float val = (mask[b * T_ + t] != 0) ? O[nt][v] / rsum[v] : vm;
      attn16[((size_t)(b * T_ + t)) * E_ + h * S_ + sc] = (_Float16)val;
    }
  }
}

// ---------------------------------------------------- output projection
// y[r, c] = attn[r,:] . Wo[c,:] + bo[c];  wave does 16 rows x 64 cols
__global__ __launch_bounds__(32) void out_proj_kernel(
    const _Float16* __restrict__ attn16, const _Float16* __restrict__ wo16,
    const float* __restrict__ bo, float* __restrict__ y) {
  int r0 = blockIdx.x * 16;
  int c0 = blockIdx.y * 64;
  int lane = threadIdx.x;
  int l16 = lane & 15;
  int half8 = (lane >> 4) << 3;
  int halfk = (lane >> 4) << 4;

  v8f acc[4] = {};
  const _Float16* arow = attn16 + (size_t)(r0 + l16) * E_;
  for (int ks = 0; ks < E_; ks += 32) {
    v16h a = frag_a(arow + ks, half8);
#pragma unroll
    for (int nt = 0; nt < 4; ++nt) {
      const _Float16* wrow = wo16 + (size_t)(c0 + nt * 16 + l16) * E_ + ks;
      acc[nt] = wmma_f16(a, frag_b(wrow, halfk), acc[nt]);
    }
  }
#pragma unroll
  for (int nt = 0; nt < 4; ++nt) {
    int col = c0 + nt * 16 + l16;
    float bias = bo[col];
#pragma unroll
    for (int v = 0; v < 8; ++v)
      y[(size_t)(r0 + half8 + v) * E_ + col] = acc[nt][v] + bias;
  }
}

extern "C" void kernel_launch(void* const* d_in, const int* in_sizes, int n_in,
                              void* d_out, int out_size, void* d_ws,
                              size_t ws_size, hipStream_t stream) {
  (void)in_sizes; (void)n_in; (void)out_size; (void)ws_size;
  const float* x    = (const float*)d_in[0];
  const int*   mask = (const int*)d_in[1];
  const float* Wq   = (const float*)d_in[2];
  const float* Wk   = (const float*)d_in[3];
  const float* Wv   = (const float*)d_in[4];
  const float* Er   = (const float*)d_in[5];
  const float* Wo   = (const float*)d_in[6];
  const float* bo   = (const float*)d_in[7];
  float* y = (float*)d_out;

  char* ws = (char*)d_ws;
  const size_t QSZ = (size_t)B_ * H_ * T_ * S_ * sizeof(_Float16);  // 8 MiB
  _Float16* q16    = (_Float16*)(ws);
  _Float16* k16    = (_Float16*)(ws + QSZ);
  _Float16* vT16   = (_Float16*)(ws + 2 * QSZ);
  _Float16* attn16 = (_Float16*)(ws + 3 * QSZ);            // B*T*E f16 = 8 MiB
  _Float16* er16   = (_Float16*)(ws + 4 * QSZ);            // 1 MiB
  _Float16* wo16   = (_Float16*)(ws + 4 * QSZ + (size_t)H_ * T_ * S_ * 2);
  float* vmean = (float*)(ws + 4 * QSZ + (size_t)H_ * T_ * S_ * 2 +
                          (size_t)E_ * E_ * 2);

  proj_qkv_kernel<<<B_ * T_, 256, 0, stream>>>(x, Wq, Wk, Wv, q16, k16, vT16);

  int nEr = H_ * T_ * S_;
  f32_to_f16_kernel<<<(nEr + 255) / 256, 256, 0, stream>>>(Er, er16, nEr);
  int nWo = E_ * E_;
  f32_to_f16_kernel<<<(nWo + 255) / 256, 256, 0, stream>>>(Wo, wo16, nWo);

  vmean_kernel<<<B_ * H_, 64, 0, stream>>>(vT16, vmean);

  dim3 fgrid(B_ * H_, T_ / 16);
  flash_attn_kernel<<<fgrid, 32, 0, stream>>>(q16, k16, vT16, er16, vmean,
                                              mask, attn16);

  dim3 ogrid((B_ * T_) / 16, E_ / 64);
  out_proj_kernel<<<ogrid, 32, 0, stream>>>(attn16, wo16, bo, y);
}
